// DebertaModel_1099511627904
// MI455X (gfx1250) — compile-verified, matrix operand order checked by reading, not run on hardware
//
#include <hip/hip_runtime.h>
#include <hip/hip_bf16.h>

// ---------------------------------------------------------------------------
// DeBERTa-v2 forward, CDNA5 (gfx1250).
// - All GEMMs: bf16 WMMA (v_wmma_f32_16x16x32_bf16), f32 accumulate.
// - Weights pre-converted fp32->bf16 and pre-transposed to [N,K] once per call.
// - Activations kept bf16 between GEMMs (LN/softmax compute f32, store bf16).
// - GEMM tiles staged with global_load_async_to_lds_b128 into double-buffered
//   LDS, synchronized with s_wait_asynccnt (ASYNCcnt) + workgroup barriers.
// ---------------------------------------------------------------------------

static constexpr int cB   = 2;
static constexpr int cS   = 512;
static constexpr int cH   = 1024;
static constexpr int cNH  = 16;
static constexpr int cD   = 64;      // H / NH
static constexpr int cL   = 4;
static constexpr int cFF  = 4096;
static constexpr int cNC  = 14;
static constexpr int cSPAN = 256;
static constexpr int cR   = 2 * cSPAN;   // 512 relative buckets
static constexpr int cBS  = cB * cS;     // 1024 token rows
static constexpr int cNCP = 128;         // decoder N padded for async loads

typedef __bf16 bf16_t;
typedef __attribute__((ext_vector_type(16))) __bf16 v16bf;
typedef __attribute__((ext_vector_type(8)))  __bf16 v8bf;
typedef __attribute__((ext_vector_type(8)))  float  v8f;

union F16u { v16bf v; v8bf h[2]; };
union F8u  { v8f v; float f[8]; };

__device__ __forceinline__ float gelu_exact(float x) {
  return 0.5f * x * (1.0f + erff(x * 0.70710678118654752f));
}

// ---------------------------------------------------------------------------
// Tiled WMMA GEMM:  C = act(A @ B + bias [+ res])
//   A: bf16 [M,K], row stride lda (M always a multiple of 128 here)
//   B: bf16 [N,K] n-major (BT=true, async path; rows n0..n0+127 must exist)
//      bf16 [K,N] k-major (BT=false, VALU transpose path, bounds-checked)
// Double-buffered LDS; A (and B when BT) staged via async-to-LDS b128.
// Block 256 threads / 8 waves, tile 128x128, K-step 32, wave tile 32x64.
// ---------------------------------------------------------------------------
template<bool BT, typename OutT>
__global__ __launch_bounds__(256)
void gemm_wmma(const bf16_t* __restrict__ A, int lda, long Ao, long Ai,
               const bf16_t* __restrict__ Bm, int ldb, long Bo, long Bi,
               OutT* __restrict__ C, int ldc, long Co, long Ci,
               const float* __restrict__ bias,
               const bf16_t* __restrict__ res, int ldr,
               int M, int N, int K, int innerCount, int act)
{
  __shared__ __align__(16) bf16_t As[2][128][40];   // row stride 80B (16B mult)
  __shared__ __align__(16) bf16_t Bs[2][128][40];

  const int z     = blockIdx.z;
  const int outer = z / innerCount;
  const int inner = z - outer * innerCount;
  A  += outer * Ao + inner * Ai;
  Bm += outer * Bo + inner * Bi;
  C  += outer * Co + inner * Ci;

  const int tid  = threadIdx.x;
  const int m0   = blockIdx.y * 128;
  const int n0   = blockIdx.x * 128;
  const int wave = tid >> 5;
  const int lane = tid & 31;
  const int wm   = (wave & 3) * 32;
  const int wn   = (wave >> 2) * 64;
  const int lr   = lane & 15;
  const int lh   = lane >> 4;

  const unsigned ldsA = (unsigned)(uintptr_t)(&As[0][0][0]);
  const unsigned ldsB = (unsigned)(uintptr_t)(&Bs[0][0][0]);
  const unsigned bufStride = 128u * 40u * 2u;       // 10240 B per buffer

  const int rowQ = tid >> 2;      // 0..63
  const int chk  = tid & 3;       // 16B chunk within a 64B row

  auto issueA = [&](int k0, int buf) {
    #pragma unroll
    for (int i = 0; i < 2; i++) {
      const int r = rowQ + 64 * i;
      const unsigned lo = ldsA + (unsigned)buf * bufStride + (unsigned)(r * 80 + chk * 16);
      const unsigned long long g =
          (unsigned long long)(const void*)(A + (long)(m0 + r) * lda + k0 + chk * 8);
      asm volatile("global_load_async_to_lds_b128 %0, %1, off"
                   :: "v"(lo), "v"(g) : "memory");
    }
  };
  auto issueB = [&](int k0, int buf) {
    if (BT) {
      #pragma unroll
      for (int i = 0; i < 2; i++) {
        const int r = rowQ + 64 * i;
        const unsigned lo = ldsB + (unsigned)buf * bufStride + (unsigned)(r * 80 + chk * 16);
        const unsigned long long g =
            (unsigned long long)(const void*)(Bm + (long)(n0 + r) * ldb + k0 + chk * 8);
        asm volatile("global_load_async_to_lds_b128 %0, %1, off"
                     :: "v"(lo), "v"(g) : "memory");
      }
    } else {
      // k-major B: transpose into LDS with bounds-checked VALU loads.
      const int bk  = tid >> 3;         // 0..31
      const int bnb = (tid & 7) << 4;   // 0..112
      const bf16_t* p = Bm + (long)(k0 + bk) * ldb + n0 + bnb;
      bf16_t t[16];
      #pragma unroll
      for (int j = 0; j < 16; j++) t[j] = (n0 + bnb + j < N) ? p[j] : (bf16_t)0.0f;
      #pragma unroll
      for (int j = 0; j < 16; j++) Bs[buf][bnb + j][bk] = t[j];
    }
  };

  F8u acc[2][4];
  #pragma unroll
  for (int i = 0; i < 2; i++)
    #pragma unroll
    for (int j = 0; j < 4; j++)
      #pragma unroll
      for (int e = 0; e < 8; e++) acc[i][j].f[e] = 0.0f;

  issueA(0, 0);
  issueB(0, 0);

  int cb = 0;
  for (int k0 = 0; k0 < K; k0 += 32) {
    const bool hasNext = (k0 + 32) < K;
    if (hasNext) {
      issueA(k0 + 32, cb ^ 1);
      issueB(k0 + 32, cb ^ 1);
      asm volatile("s_wait_asynccnt %0" :: "n"(BT ? 4 : 2) : "memory");
    } else {
      asm volatile("s_wait_asynccnt 0" ::: "memory");
    }
    __syncthreads();

    F16u fa[2], fb[4];
    #pragma unroll
    for (int mt = 0; mt < 2; mt++) {
      const bf16_t* p = &As[cb][wm + mt * 16 + lr][lh * 8];
      fa[mt].h[0] = *(const v8bf*)p;
      fa[mt].h[1] = *(const v8bf*)(p + 16);
    }
    #pragma unroll
    for (int nt = 0; nt < 4; nt++) {
      const bf16_t* p = &Bs[cb][wn + nt * 16 + lr][lh * 8];
      fb[nt].h[0] = *(const v8bf*)p;
      fb[nt].h[1] = *(const v8bf*)(p + 16);
    }
    #pragma unroll
    for (int mt = 0; mt < 2; mt++)
      #pragma unroll
      for (int nt = 0; nt < 4; nt++)
        acc[mt][nt].v = __builtin_amdgcn_wmma_f32_16x16x32_bf16(
            false, fa[mt].v, false, fb[nt].v,
            (short)0, acc[mt][nt].v, false, false);

    __syncthreads();
    cb ^= 1;
  }

  #pragma unroll
  for (int mt = 0; mt < 2; mt++) {
    #pragma unroll
    for (int nt = 0; nt < 4; nt++) {
      const int col = n0 + wn + nt * 16 + lr;
      if (col >= N) continue;
      const float bv = bias ? bias[col] : 0.0f;
      #pragma unroll
      for (int e = 0; e < 8; e++) {
        const int row = m0 + wm + mt * 16 + lh * 8 + e;
        if (row >= M) continue;
        float c = acc[mt][nt].f[e] + bv;
        if (res) c += (float)res[(long)row * ldr + col];
        if (act == 1) c = gelu_exact(c);
        C[(long)row * ldc + col] = (OutT)c;
      }
    }
  }
}

// ---------------------------------------------------------------------------
// Weight prep: W [K,N] fp32 -> WT [Npad,K] bf16 (LDS-tiled transpose,
// zero-fill rows n in [N, Npad)).
// ---------------------------------------------------------------------------
__global__ __launch_bounds__(256)
void transpose_cvt(const float* __restrict__ W, bf16_t* __restrict__ WT,
                   int Kd, int Nd, int Npad)
{
  __shared__ float t[32][33];
  const int kb = blockIdx.y * 32, nb = blockIdx.x * 32;
  const int tx = threadIdx.x & 31, ty = threadIdx.x >> 5;   // 32 x 8
  #pragma unroll
  for (int i = 0; i < 32; i += 8) {
    const int k = kb + ty + i, n = nb + tx;
    t[ty + i][tx] = (k < Kd && n < Nd) ? W[(long)k * Nd + n] : 0.0f;
  }
  __syncthreads();
  #pragma unroll
  for (int i = 0; i < 32; i += 8) {
    const int n = nb + ty + i, k = kb + tx;
    if (n < Npad && k < Kd) WT[(long)n * Kd + k] = (bf16_t)t[tx][ty + i];
  }
}

// ---------------------------------------------------------------------------
// LayerNorm (fp32 in -> bf16 out), rows of length H=1024.
// ---------------------------------------------------------------------------
__global__ __launch_bounds__(256)
void layernorm_k(const float* __restrict__ x, const float* __restrict__ s,
                 const float* __restrict__ bvec, bf16_t* __restrict__ y)
{
  const int row = blockIdx.x;
  const int tid = threadIdx.x;
  const float* xr = x + (long)row * cH;
  float v[4];
  float sum = 0.0f;
  #pragma unroll
  for (int i = 0; i < 4; i++) { v[i] = xr[tid + i * 256]; sum += v[i]; }
  __shared__ float red[256];
  red[tid] = sum; __syncthreads();
  for (int o = 128; o > 0; o >>= 1) { if (tid < o) red[tid] += red[tid + o]; __syncthreads(); }
  const float mu = red[0] * (1.0f / cH);
  __syncthreads();
  float sq = 0.0f;
  #pragma unroll
  for (int i = 0; i < 4; i++) { float d = v[i] - mu; sq += d * d; }
  red[tid] = sq; __syncthreads();
  for (int o = 128; o > 0; o >>= 1) { if (tid < o) red[tid] += red[tid + o]; __syncthreads(); }
  const float rstd = rsqrtf(red[0] * (1.0f / cH) + 1e-7f);
  #pragma unroll
  for (int i = 0; i < 4; i++) {
    const int c = tid + i * 256;
    y[(long)row * cH + c] = (bf16_t)((v[i] - mu) * rstd * s[c] + bvec[c]);
  }
}

// Embedding gather + LayerNorm + attention-mask multiply (bf16 out).
__global__ __launch_bounds__(256)
void embed_ln(const float* __restrict__ emb, const int* __restrict__ ids,
              const int* __restrict__ amask, const float* __restrict__ s,
              const float* __restrict__ bvec, bf16_t* __restrict__ y)
{
  const int row = blockIdx.x;
  const int tid = threadIdx.x;
  const float* xr = emb + (long)ids[row] * cH;
  const float mk = amask[row] ? 1.0f : 0.0f;
  float v[4];
  float sum = 0.0f;
  #pragma unroll
  for (int i = 0; i < 4; i++) { v[i] = xr[tid + i * 256]; sum += v[i]; }
  __shared__ float red[256];
  red[tid] = sum; __syncthreads();
  for (int o = 128; o > 0; o >>= 1) { if (tid < o) red[tid] += red[tid + o]; __syncthreads(); }
  const float mu = red[0] * (1.0f / cH);
  __syncthreads();
  float sq = 0.0f;
  #pragma unroll
  for (int i = 0; i < 4; i++) { float d = v[i] - mu; sq += d * d; }
  red[tid] = sq; __syncthreads();
  for (int o = 128; o > 0; o >>= 1) { if (tid < o) red[tid] += red[tid + o]; __syncthreads(); }
  const float rstd = rsqrtf(red[0] * (1.0f / cH) + 1e-7f);
  #pragma unroll
  for (int i = 0; i < 4; i++) {
    const int c = tid + i * 256;
    y[(long)row * cH + c] = (bf16_t)(((v[i] - mu) * rstd * s[c] + bvec[c]) * mk);
  }
}

// DeBERTa-v2 log-bucket relative position indices.
__global__ void rel_buckets(int* __restrict__ c2pIdx, int* __restrict__ p2cIdx)
{
  const int idx = blockIdx.x * 256 + threadIdx.x;
  if (idx >= cS * cS) return;
  const int q = idx / cS, k = idx - q * cS;
  const int rel = q - k;
  const int mid = 128;  // BUCKET/2
  const int a = (rel < mid && rel > -mid) ? (mid - 1) : (rel < 0 ? -rel : rel);
  const float logpos = ceilf(logf((float)a / (float)mid) /
                             logf((float)(512 - 1) / (float)mid) * (float)(mid - 1)) + (float)mid;
  const float sgn = (rel > 0) ? 1.0f : ((rel < 0) ? -1.0f : 0.0f);
  const int bucket = (a <= mid) ? rel : (int)(logpos * sgn);
  int c = bucket + cSPAN;  c = c < 0 ? 0 : (c > 2 * cSPAN - 1 ? 2 * cSPAN - 1 : c);
  int p = -bucket + cSPAN; p = p < 0 ? 0 : (p > 2 * cSPAN - 1 ? 2 * cSPAN - 1 : p);
  c2pIdx[idx] = c;
  p2cIdx[idx] = p;
}

// Disentangled bias gather + scale + mask + softmax; fp32 scores -> bf16 probs.
__global__ __launch_bounds__(256)
void attn_softmax(const float* __restrict__ scores, const bf16_t* __restrict__ c2p,
                  const bf16_t* __restrict__ p2c, const int* __restrict__ c2pIdx,
                  const int* __restrict__ p2cIdx, const int* __restrict__ amask,
                  bf16_t* __restrict__ probs, float invScale)
{
  const int q = blockIdx.x, h = blockIdx.y, b = blockIdx.z;
  const int tid = threadIdx.x;
  const long bh = (long)(b * cNH + h);
  const float* row = scores + (bh * cS + q) * cS;
  const bf16_t* c2pr = c2p + (bh * cS + q) * cR;
  bf16_t* prow = probs + (bh * cS + q) * cS;
  const bool qm = amask[b * cS + q] != 0;

  float vals[2]; bool ms[2];
  #pragma unroll
  for (int t = 0; t < 2; t++) {
    const int j = tid + t * 256;
    float sc = row[j]
             + (float)c2pr[c2pIdx[q * cS + j]]
             + (float)p2c[(bh * cS + j) * cR + p2cIdx[j * cS + q]];
    sc *= invScale;
    const bool m = qm && (amask[b * cS + j] != 0);
    ms[t] = m;
    vals[t] = m ? sc : -3.402823466e38f;
  }
  __shared__ float red[256];
  red[tid] = fmaxf(vals[0], vals[1]); __syncthreads();
  for (int o = 128; o > 0; o >>= 1) { if (tid < o) red[tid] = fmaxf(red[tid], red[tid + o]); __syncthreads(); }
  const float mx = red[0];
  __syncthreads();
  float e[2];
  e[0] = expf(vals[0] - mx);
  e[1] = expf(vals[1] - mx);
  red[tid] = e[0] + e[1]; __syncthreads();
  for (int o = 128; o > 0; o >>= 1) { if (tid < o) red[tid] += red[tid + o]; __syncthreads(); }
  const float inv = 1.0f / red[0];
  #pragma unroll
  for (int t = 0; t < 2; t++) {
    const int j = tid + t * 256;
    prow[j] = (bf16_t)(ms[t] ? e[t] * inv : 0.0f);
  }
}

// Per-row masked NLL over NC=14 logits (logits fp32 in d_out).
__global__ void nll_rows(const float* __restrict__ logits, const int* __restrict__ labels,
                         const int* __restrict__ amask, float* __restrict__ rownll,
                         float* __restrict__ rowmask)
{
  const int row = blockIdx.x * 256 + threadIdx.x;
  if (row >= cBS) return;
  const float* lg = logits + (long)row * cNC;
  float mx = lg[0];
  #pragma unroll
  for (int c = 1; c < cNC; c++) mx = fmaxf(mx, lg[c]);
  float s = 0.0f;
  #pragma unroll
  for (int c = 0; c < cNC; c++) s += expf(lg[c] - mx);
  const float lse = logf(s) + mx;
  const float m = amask[row] ? 1.0f : 0.0f;
  rownll[row] = (lse - lg[labels[row]]) * m;
  rowmask[row] = m;
}

__global__ __launch_bounds__(256)
void loss_reduce(const float* __restrict__ rownll, const float* __restrict__ rowmask,
                 float* __restrict__ outLoss)
{
  const int tid = threadIdx.x;
  float sn = 0.0f, sm = 0.0f;
  for (int i = tid; i < cBS; i += 256) { sn += rownll[i]; sm += rowmask[i]; }
  __shared__ float rn[256], rm[256];
  rn[tid] = sn; rm[tid] = sm; __syncthreads();
  for (int o = 128; o > 0; o >>= 1) {
    if (tid < o) { rn[tid] += rn[tid + o]; rm[tid] += rm[tid + o]; }
    __syncthreads();
  }
  if (tid == 0) outLoss[0] = rn[0] / fmaxf(rm[0], 1.0f);
}

// ---------------------------------------------------------------------------
// Host-side dispatchers
// ---------------------------------------------------------------------------
template<typename OutT>
static inline void gemm(hipStream_t st, bool bT,
                        const bf16_t* A, int lda, long Ao, long Ai,
                        const bf16_t* Bp, int ldb, long Bo, long Bi,
                        OutT* C, int ldc, long Co, long Ci,
                        const float* bias, const bf16_t* res, int ldr,
                        int M, int N, int K, int batch, int innerCount, int act)
{
  dim3 g((N + 127) / 128, (M + 127) / 128, batch);
  if (bT)
    gemm_wmma<true, OutT><<<g, 256, 0, st>>>(A, lda, Ao, Ai, Bp, ldb, Bo, Bi,
                                             C, ldc, Co, Ci, bias, res, ldr,
                                             M, N, K, innerCount, act);
  else
    gemm_wmma<false, OutT><<<g, 256, 0, st>>>(A, lda, Ao, Ai, Bp, ldb, Bo, Bi,
                                              C, ldc, Co, Ci, bias, res, ldr,
                                              M, N, K, innerCount, act);
}

static inline void tcvt(hipStream_t st, const float* W, bf16_t* WT,
                        int Kd, int Nd, int Npad)
{
  dim3 g((Npad + 31) / 32, (Kd + 31) / 32, 1);
  transpose_cvt<<<g, 256, 0, st>>>(W, WT, Kd, Nd, Npad);
}

extern "C" void kernel_launch(void* const* d_in, const int* in_sizes, int n_in,
                              void* d_out, int out_size, void* d_ws, size_t ws_size,
                              hipStream_t stream) {
  (void)in_sizes; (void)n_in; (void)out_size; (void)ws_size;

  const float* word_emb = (const float*)d_in[0];
  const float* emb_ln_s = (const float*)d_in[1];
  const float* emb_ln_b = (const float*)d_in[2];
  const float* rel_emb  = (const float*)d_in[3];
  const float* rel_ln_s = (const float*)d_in[4];
  const float* rel_ln_b = (const float*)d_in[5];
  const float* Wq = (const float*)d_in[6];
  const float* bq = (const float*)d_in[7];
  const float* Wk = (const float*)d_in[8];
  const float* bk = (const float*)d_in[9];
  const float* Wv = (const float*)d_in[10];
  const float* bv = (const float*)d_in[11];
  const float* Wo = (const float*)d_in[12];
  const float* bo = (const float*)d_in[13];
  const float* ln1_s = (const float*)d_in[14];
  const float* ln1_b = (const float*)d_in[15];
  const float* W1 = (const float*)d_in[16];
  const float* b1 = (const float*)d_in[17];
  const float* W2 = (const float*)d_in[18];
  const float* b2 = (const float*)d_in[19];
  const float* ln2_s = (const float*)d_in[20];
  const float* ln2_b = (const float*)d_in[21];
  const float* Wt = (const float*)d_in[22];
  const float* bt = (const float*)d_in[23];
  const float* tln_s = (const float*)d_in[24];
  const float* tln_b = (const float*)d_in[25];
  const float* Wd = (const float*)d_in[26];
  const float* bd = (const float*)d_in[27];
  const int* input_ids = (const int*)d_in[28];
  const int* amask     = (const int*)d_in[29];
  const int* labels    = (const int*)d_in[30];

  float* out = (float*)d_out;

  // --- workspace layout (256B-aligned chunks) ---
  char* base = (char*)d_ws;
  size_t off = 0;
  auto allocF = [&](size_t n) -> float* {
    float* p = (float*)(base + off);
    off = (off + n * sizeof(float) + 255) & ~(size_t)255;
    return p;
  };
  auto allocH = [&](size_t n) -> bf16_t* {
    bf16_t* p = (bf16_t*)(base + off);
    off = (off + n * sizeof(bf16_t) + 255) & ~(size_t)255;
    return p;
  };
  auto allocI = [&](size_t n) -> int* {
    int* p = (int*)(base + off);
    off = (off + n * sizeof(int) + 255) & ~(size_t)255;
    return p;
  };

  // fp32 buffers
  float* tmp    = allocF((size_t)cBS * cH);
  float* scores = allocF((size_t)cB * cNH * cS * cS);
  float* rownll  = allocF(cBS);
  float* rowmask = allocF(cBS);
  int* c2pIdx = allocI((size_t)cS * cS);
  int* p2cIdx = allocI((size_t)cS * cS);
  // bf16 activations
  bf16_t* hbuf   = allocH((size_t)cBS * cH);
  bf16_t* qb     = allocH((size_t)cBS * cH);
  bf16_t* kb     = allocH((size_t)cBS * cH);
  bf16_t* vb     = allocH((size_t)cBS * cH);
  bf16_t* ctx    = allocH((size_t)cBS * cH);
  bf16_t* tbuf   = allocH((size_t)cBS * cH);
  bf16_t* relLN  = allocH((size_t)cR * cH);
  bf16_t* posK   = allocH((size_t)cR * cH);
  bf16_t* posQ   = allocH((size_t)cR * cH);
  bf16_t* probsb = allocH((size_t)cB * cNH * cS * cS);
  bf16_t* c2pb   = allocH((size_t)cB * cNH * cS * cR);
  bf16_t* p2cb   = allocH((size_t)cB * cNH * cS * cR);
  bf16_t* ffb    = allocH((size_t)cBS * cFF);
  // bf16 transposed weights [N,K]
  bf16_t* WqT = allocH((size_t)cL * cH * cH);
  bf16_t* WkT = allocH((size_t)cL * cH * cH);
  bf16_t* WvT = allocH((size_t)cL * cH * cH);
  bf16_t* WoT = allocH((size_t)cL * cH * cH);
  bf16_t* W1T = allocH((size_t)cL * cFF * cH);   // [FF,H] per layer
  bf16_t* W2T = allocH((size_t)cL * cH * cFF);   // [H,FF] per layer
  bf16_t* WtT = allocH((size_t)cH * cH);
  bf16_t* WdT = allocH((size_t)cNCP * cH);       // padded to 128 rows

  const float invScale = 1.0f / sqrtf((float)(cD * 3));

  // --- weight prep: fp32 [K,N] -> bf16 [N,K] ---
  for (int l = 0; l < cL; l++) {
    tcvt(stream, Wq + (size_t)l * cH * cH, WqT + (size_t)l * cH * cH, cH, cH, cH);
    tcvt(stream, Wk + (size_t)l * cH * cH, WkT + (size_t)l * cH * cH, cH, cH, cH);
    tcvt(stream, Wv + (size_t)l * cH * cH, WvT + (size_t)l * cH * cH, cH, cH, cH);
    tcvt(stream, Wo + (size_t)l * cH * cH, WoT + (size_t)l * cH * cH, cH, cH, cH);
    tcvt(stream, W1 + (size_t)l * cH * cFF, W1T + (size_t)l * cFF * cH, cH, cFF, cFF);
    tcvt(stream, W2 + (size_t)l * cFF * cH, W2T + (size_t)l * cH * cFF, cFF, cH, cH);
  }
  tcvt(stream, Wt, WtT, cH, cH, cH);
  tcvt(stream, Wd, WdT, cH, cNC, cNCP);

  // --- prologue ---
  rel_buckets<<<(cS * cS + 255) / 256, 256, 0, stream>>>(c2pIdx, p2cIdx);
  embed_ln<<<cBS, 256, 0, stream>>>(word_emb, input_ids, amask, emb_ln_s, emb_ln_b, hbuf);
  layernorm_k<<<cR, 256, 0, stream>>>(rel_emb, rel_ln_s, rel_ln_b, relLN);

  // --- transformer layers ---
  for (int l = 0; l < cL; l++) {
    const bf16_t* wqT = WqT + (size_t)l * cH * cH;
    const bf16_t* wkT = WkT + (size_t)l * cH * cH;
    const bf16_t* wvT = WvT + (size_t)l * cH * cH;
    const bf16_t* woT = WoT + (size_t)l * cH * cH;
    const bf16_t* w1T = W1T + (size_t)l * cFF * cH;
    const bf16_t* w2T = W2T + (size_t)l * cH * cFF;
    const float* bql = bq + (size_t)l * cH;
    const float* bkl = bk + (size_t)l * cH;
    const float* bvl = bv + (size_t)l * cH;
    const float* bol = bo + (size_t)l * cH;
    const float* b1l = b1 + (size_t)l * cFF;
    const float* b2l = b2 + (size_t)l * cH;
    const float* l1s = ln1_s + (size_t)l * cH;
    const float* l1b = ln1_b + (size_t)l * cH;
    const float* l2s = ln2_s + (size_t)l * cH;
    const float* l2b = ln2_b + (size_t)l * cH;

    // Q, K, V projections: bf16 out
    gemm<bf16_t>(stream, true, hbuf, cH, 0, 0, wqT, cH, 0, 0, qb, cH, 0, 0,
                 bql, nullptr, 0, cBS, cH, cH, 1, 1, 0);
    gemm<bf16_t>(stream, true, hbuf, cH, 0, 0, wkT, cH, 0, 0, kb, cH, 0, 0,
                 bkl, nullptr, 0, cBS, cH, cH, 1, 1, 0);
    gemm<bf16_t>(stream, true, hbuf, cH, 0, 0, wvT, cH, 0, 0, vb, cH, 0, 0,
                 bvl, nullptr, 0, cBS, cH, cH, 1, 1, 0);
    // shared-key position projections
    gemm<bf16_t>(stream, true, relLN, cH, 0, 0, wkT, cH, 0, 0, posK, cH, 0, 0,
                 bkl, nullptr, 0, cR, cH, cH, 1, 1, 0);
    gemm<bf16_t>(stream, true, relLN, cH, 0, 0, wqT, cH, 0, 0, posQ, cH, 0, 0,
                 bql, nullptr, 0, cR, cH, cH, 1, 1, 0);

    // scores = Q . K^T (batched B*NH; K rows are n-major -> BT async path)
    gemm<float>(stream, true, qb, cH, (long)cS * cH, cD, kb, cH, (long)cS * cH, cD,
                scores, cS, (long)cNH * cS * cS, (long)cS * cS, nullptr, nullptr, 0,
                cS, cS, cD, cB * cNH, cNH, 0);
    // c2p = Q . posK^T
    gemm<bf16_t>(stream, true, qb, cH, (long)cS * cH, cD, posK, cH, 0, cD,
                 c2pb, cR, (long)cNH * cS * cR, (long)cS * cR, nullptr, nullptr, 0,
                 cS, cR, cD, cB * cNH, cNH, 0);
    // p2c = K . posQ^T
    gemm<bf16_t>(stream, true, kb, cH, (long)cS * cH, cD, posQ, cH, 0, cD,
                 p2cb, cR, (long)cNH * cS * cR, (long)cS * cR, nullptr, nullptr, 0,
                 cS, cR, cD, cB * cNH, cNH, 0);

    attn_softmax<<<dim3(cS, cNH, cB), 256, 0, stream>>>(scores, c2pb, p2cb,
                                                        c2pIdx, p2cIdx, amask,
                                                        probsb, invScale);

    // ctx = P @ V (V is k-major: VALU transpose path)
    gemm<bf16_t>(stream, false, probsb, cS, (long)cNH * cS * cS, (long)cS * cS,
                 vb, cH, (long)cS * cH, cD,
                 ctx, cH, (long)cS * cH, cD, nullptr, nullptr, 0,
                 cS, cD, cS, cB * cNH, cNH, 0);

    // out projection + residual -> fp32 tmp, then LN1 -> bf16 h
    gemm<float>(stream, true, ctx, cH, 0, 0, woT, cH, 0, 0, tmp, cH, 0, 0,
                bol, hbuf, cH, cBS, cH, cH, 1, 1, 0);
    layernorm_k<<<cBS, 256, 0, stream>>>(tmp, l1s, l1b, hbuf);

    // FFN
    gemm<bf16_t>(stream, true, hbuf, cH, 0, 0, w1T, cH, 0, 0, ffb, cFF, 0, 0,
                 b1l, nullptr, 0, cBS, cFF, cH, 1, 1, 1);
    gemm<float>(stream, true, ffb, cFF, 0, 0, w2T, cFF, 0, 0, tmp, cH, 0, 0,
                b2l, hbuf, cH, cBS, cH, cFF, 1, 1, 0);
    layernorm_k<<<cBS, 256, 0, stream>>>(tmp, l2s, l2b, hbuf);
  }

  // --- transform head ---
  gemm<float>(stream, true, hbuf, cH, 0, 0, WtT, cH, 0, 0, tmp, cH, 0, 0,
              bt, nullptr, 0, cBS, cH, cH, 1, 1, 1);
  layernorm_k<<<cBS, 256, 0, stream>>>(tmp, tln_s, tln_b, tbuf);

  // --- decoder -> logits (fp32, directly into d_out) ---
  gemm<float>(stream, true, tbuf, cH, 0, 0, WdT, cH, 0, 0, out, cNC, 0, 0,
              bd, nullptr, 0, cBS, cNC, cH, 1, 1, 0);

  // --- masked cross-entropy loss -> d_out[B*S*NC] ---
  nll_rows<<<(cBS + 255) / 256, 256, 0, stream>>>(out, labels, amask, rownll, rowmask);
  loss_reduce<<<1, 256, 0, stream>>>(rownll, rowmask, out + (size_t)cBS * cNC);
}
